// MIKT_80358838108833
// MI455X (gfx1250) — matrix operand order repaired and checked by reading, not compiled, not run on hardware
//
#include <hip/hip_runtime.h>
#include <hip/hip_bf16.h>
#include <stdint.h>

// ---------------- types / helpers ----------------
typedef __attribute__((ext_vector_type(16))) __bf16        v16bf;
typedef __attribute__((ext_vector_type(16))) unsigned short v16us;
typedef __attribute__((ext_vector_type(8)))  float          v8f;

constexpr int Bc = 32, Sc = 512, Dc = 512, Hc = 8, FFc = 2048, NBc = 2, DKc = 64;
constexpr int Mrows = Bc * Sc; // 16384

static __device__ __forceinline__ unsigned short f2bf(float f) {
  unsigned int u = __builtin_bit_cast(unsigned int, f);
  u += 0x7fffu + ((u >> 16) & 1u); // RNE
  return (unsigned short)(u >> 16);
}
static __device__ __forceinline__ unsigned int pack2bf(float a, float b) {
  return (unsigned int)f2bf(a) | ((unsigned int)f2bf(b) << 16);
}

static __device__ __forceinline__ v8f wmma_bf16(v16us a, v16us b, v8f c) {
  return __builtin_amdgcn_wmma_f32_16x16x32_bf16(
      false, __builtin_bit_cast(v16bf, a),
      false, __builtin_bit_cast(v16bf, b),
      (short)0, c, false, false);
}

// Async global->LDS B128 copy (CDNA5, tracked by ASYNCcnt).
// LDS byte address is the low 32 bits of the generic pointer (LDS aperture).
static __device__ __forceinline__ void async_b128(const void* gptr, void* lptr) {
  unsigned lds = (unsigned)(uintptr_t)lptr;
  unsigned long long ga = (unsigned long long)(uintptr_t)gptr;
  asm volatile("global_load_async_to_lds_b128 %0, %1, off"
               :: "v"(lds), "v"(ga) : "memory");
}
static __device__ __forceinline__ void wait_async0() {
  asm volatile("s_wait_asynccnt 0x0" ::: "memory");
}

// A-matrix 16x32 bf16 layout: lanes 0-15 hold K={0..7,16..23}, lanes 16-31 K={8..15,24..31}
// VGPR v holds pair at kb = 2v + (v>=4?8:0) + 8*khalf (pairs contiguous in K)
static __device__ __forceinline__ v16us frag_apairs(const unsigned short* rowptr, int khalf) {
  v16us r;
#pragma unroll
  for (int v = 0; v < 8; ++v) {
    int kb = 2 * v + ((v >= 4) ? 8 : 0) + 8 * khalf;
    unsigned int pair = *reinterpret_cast<const unsigned int*>(rowptr + kb);
    r[2 * v]     = (unsigned short)(pair & 0xffffu);
    r[2 * v + 1] = (unsigned short)(pair >> 16);
  }
  return r;
}
// B-matrix 32x16 bf16 layout: lanes 0-15 hold K={0..7,16..23}, lanes 16-31 K={8..15,24..31} (same as A)
static __device__ __forceinline__ v16us frag_bpairs(const unsigned short* colptr, int khalf) {
  v16us r;
#pragma unroll
  for (int v = 0; v < 8; ++v) {
    int kb = 2 * v + ((v >= 4) ? 8 : 0) + 8 * khalf;
    unsigned int pair = *reinterpret_cast<const unsigned int*>(colptr + kb);
    r[2 * v]     = (unsigned short)(pair & 0xffffu);
    r[2 * v + 1] = (unsigned short)(pair >> 16);
  }
  return r;
}

// ---------------- prep: x = q+pe, y = qa+pe ----------------
__global__ __launch_bounds__(256) void prep_kernel(
    const float* __restrict__ qe, const float* __restrict__ qa,
    const float* __restrict__ pe, float* __restrict__ xf,
    unsigned short* __restrict__ xb, unsigned short* __restrict__ yb) {
  int i4 = blockIdx.x * 256 + threadIdx.x;
  int e = i4 * 4;
  int pei = e & (Sc * Dc - 1); // S*D is a power of two
  float4 qv = *(const float4*)(qe + e);
  float4 av = *(const float4*)(qa + e);
  float4 pv = *(const float4*)(pe + pei);
  float4 xv = make_float4(qv.x + pv.x, qv.y + pv.y, qv.z + pv.z, qv.w + pv.w);
  float4 yv = make_float4(av.x + pv.x, av.y + pv.y, av.z + pv.z, av.w + pv.w);
  *(float4*)(xf + e) = xv;
  *(uint2*)(xb + e) = make_uint2(pack2bf(xv.x, xv.y), pack2bf(xv.z, xv.w));
  *(uint2*)(yb + e) = make_uint2(pack2bf(yv.x, yv.y), pack2bf(yv.z, yv.w));
}

// ---------------- fp32 -> bf16 tensor convert ----------------
__global__ __launch_bounds__(256) void cvt_bf16_kernel(
    const float* __restrict__ in, unsigned short* __restrict__ out) {
  int i4 = blockIdx.x * 256 + threadIdx.x;
  float4 v = *(const float4*)(in + (size_t)i4 * 4);
  *(uint2*)(out + (size_t)i4 * 4) = make_uint2(pack2bf(v.x, v.y), pack2bf(v.z, v.w));
}

// ---------------- GEMM: C[M,N] = A[M,K](bf16) @ W[K,N](bf16) + bias ----------------
// block tile 128x64, BK=32, 8 waves each doing 32x32 via 2x2 WMMA tiles
__global__ __launch_bounds__(256) void gemm_kernel(
    const unsigned short* __restrict__ A, const unsigned short* __restrict__ W,
    const float* __restrict__ bias, float* __restrict__ outF,
    unsigned short* __restrict__ outB, int relu, int M, int N, int K) {
  __shared__ unsigned short As[128][40]; // [m][k], 80B row stride (16B aligned)
  __shared__ unsigned short Bs[64][36];  // [n][k] (transposed in LDS)
  const int tid = threadIdx.x;
  const int lane = tid & 31, wave = tid >> 5;
  const int wm = wave & 3, wn = wave >> 2;
  const int hh = lane >> 4, nl = lane & 15;
  const int bm = blockIdx.y, bn = blockIdx.x;
  (void)M;

  v8f acc[2][2];
#pragma unroll
  for (int a = 0; a < 2; ++a)
#pragma unroll
    for (int b = 0; b < 2; ++b)
#pragma unroll
      for (int r = 0; r < 8; ++r) acc[a][b][r] = 0.0f;

  for (int k0 = 0; k0 < K; k0 += 32) {
    __syncthreads();
    // A tile 128x32 : async global->LDS (ASYNCcnt path)
#pragma unroll
    for (int it = 0; it < 2; ++it) {
      int lin = tid * 2 + it; // 0..511
      int r = lin >> 2, q = lin & 3;
      async_b128(A + (size_t)(bm * 128 + r) * K + k0 + q * 8, &As[r][q * 8]);
    }
    // W tile 32x64, stored transposed [n][k]
    {
      int k = tid >> 3, seg = tid & 7;
      uint4 d = *(const uint4*)(W + (size_t)(k0 + k) * N + bn * 64 + seg * 8);
      unsigned int dd[4] = {d.x, d.y, d.z, d.w};
#pragma unroll
      for (int j = 0; j < 4; ++j) {
        Bs[seg * 8 + j * 2][k]     = (unsigned short)(dd[j] & 0xffffu);
        Bs[seg * 8 + j * 2 + 1][k] = (unsigned short)(dd[j] >> 16);
      }
    }
    if (k0 + 32 < K) {
      __builtin_prefetch(A + (size_t)(bm * 128 + (tid >> 1)) * K + k0 + 32, 0, 1);
      __builtin_prefetch(W + (size_t)(k0 + 32 + (tid >> 3)) * N + bn * 64, 0, 1);
    }
    wait_async0();
    __syncthreads();

    v16us af[2], bf[2];
#pragma unroll
    for (int mt = 0; mt < 2; ++mt)
      af[mt] = frag_apairs(&As[wm * 32 + mt * 16 + nl][0], hh);
#pragma unroll
    for (int nt = 0; nt < 2; ++nt)
      bf[nt] = frag_bpairs(&Bs[wn * 32 + nt * 16 + nl][0], hh);
#pragma unroll
    for (int mt = 0; mt < 2; ++mt)
#pragma unroll
      for (int nt = 0; nt < 2; ++nt)
        acc[mt][nt] = wmma_bf16(af[mt], bf[nt], acc[mt][nt]);
  }

  // epilogue: C/D layout -> lane holds (m = r + 8*hh, n = nl)
#pragma unroll
  for (int mt = 0; mt < 2; ++mt)
#pragma unroll
    for (int nt = 0; nt < 2; ++nt) {
      int gn = bn * 64 + wn * 32 + nt * 16 + nl;
      float bia = bias[gn];
#pragma unroll
      for (int r = 0; r < 8; ++r) {
        int gm = bm * 128 + wm * 32 + mt * 16 + hh * 8 + r;
        float v = acc[mt][nt][r] + bia;
        if (relu) v = fmaxf(v, 0.0f);
        size_t o = (size_t)gm * N + gn;
        if (outF) outF[o] = v;
        if (outB) outB[o] = f2bf(v);
      }
    }
}

// ---------------- flash attention (q==k), causal strict-lower, forget scale ----------------
// grid (B*H, S/64), 128 threads (4 waves x 16 query rows)
__global__ __launch_bounds__(128) void attn_kernel(
    const unsigned short* __restrict__ Q, const unsigned short* __restrict__ V,
    const float* __restrict__ forget, unsigned short* __restrict__ O) {
  __shared__ unsigned short Qs[64][72];       // [qrow][dk], 144B stride (16B aligned)
  __shared__ unsigned short Ks[64][72];       // [key][dk]
  __shared__ unsigned short Vst[64][72];      // [dk][key] (transposed)
  __shared__ unsigned short Ps[4][16][72];    // per-wave P tile [m][key]

  const int tid = threadIdx.x;
  const int lane = tid & 31, wave = tid >> 5;
  const int hh = lane >> 4, nl = lane & 15;
  const int bh = blockIdx.x;
  const int b = bh >> 3, h = bh & 7;
  const int q0 = blockIdx.y * 64;
  const size_t baseQ = ((size_t)b * Sc) * Dc + (size_t)h * DKc;

  { // load Q tile (64x64) via async copies
    int r = tid >> 1, half = (tid & 1) * 32;
    const unsigned short* src = Q + baseQ + (size_t)(q0 + r) * Dc + half;
#pragma unroll
    for (int j = 0; j < 4; ++j)
      async_b128(src + j * 8, &Qs[r][half + j * 8]);
  }

  float rmax[8], rsum[8], fr[8];
  int qrow[8];
  v8f o[4];
#pragma unroll
  for (int r = 0; r < 8; ++r) {
    rmax[r] = -3.0e38f;
    rsum[r] = 0.0f;
    qrow[r] = q0 + wave * 16 + hh * 8 + r;
    fr[r] = 0.125f * forget[b * Sc + qrow[r]]; // 1/sqrt(64) * forget_rate
  }
#pragma unroll
  for (int dt = 0; dt < 4; ++dt)
#pragma unroll
    for (int r = 0; r < 8; ++r) o[dt][r] = 0.0f;

  const int nblocks = (q0 >> 6) + 1;
  for (int jb = 0; jb < nblocks; ++jb) {
    const int j0 = jb * 64;
    __syncthreads();
    { // K tile async; V tile manual (transposed scatter)
      int r = tid >> 1, half = (tid & 1) * 32;
      const unsigned short* ks = Q + baseQ + (size_t)(j0 + r) * Dc + half;
#pragma unroll
      for (int j = 0; j < 4; ++j)
        async_b128(ks + j * 8, &Ks[r][half + j * 8]);
      const unsigned short* vs = V + baseQ + (size_t)(j0 + r) * Dc + half;
#pragma unroll
      for (int j = 0; j < 4; ++j) {
        uint4 d = *(const uint4*)(vs + j * 8);
        unsigned int dd[4] = {d.x, d.y, d.z, d.w};
#pragma unroll
        for (int p = 0; p < 4; ++p) {
          Vst[half + j * 8 + p * 2][r]     = (unsigned short)(dd[p] & 0xffffu);
          Vst[half + j * 8 + p * 2 + 1][r] = (unsigned short)(dd[p] >> 16);
        }
      }
    }
    wait_async0();
    __syncthreads();

    // S = Q Kt : 4 n-tiles x 2 k-chunks
    v8f sf[4];
#pragma unroll
    for (int nt = 0; nt < 4; ++nt) {
      v8f acc;
#pragma unroll
      for (int r = 0; r < 8; ++r) acc[r] = 0.0f;
#pragma unroll
      for (int kc = 0; kc < 2; ++kc) {
        v16us a  = frag_apairs(&Qs[wave * 16 + nl][kc * 32], hh);
        v16us bb = frag_bpairs(&Ks[nt * 16 + nl][kc * 32], hh);
        acc = wmma_bf16(a, bb, acc);
      }
      sf[nt] = acc;
    }

    // online softmax, row stats via shfl within each 16-lane half
#pragma unroll
    for (int r = 0; r < 8; ++r) {
      float mx = -3.0e38f;
#pragma unroll
      for (int nt = 0; nt < 4; ++nt) {
        int key = j0 + nt * 16 + nl;
        float s = sf[nt][r] * fr[r];
        s = (key < qrow[r]) ? s : -1.0e30f; // strictly-lower causal mask
        sf[nt][r] = s;
        mx = fmaxf(mx, s);
      }
#pragma unroll
      for (int d = 1; d < 16; d <<= 1) mx = fmaxf(mx, __shfl_xor(mx, d, 32));
      float mnew = fmaxf(rmax[r], mx);
      float al = __expf(rmax[r] - mnew);
      rmax[r] = mnew;
      float ps = 0.0f;
#pragma unroll
      for (int nt = 0; nt < 4; ++nt) {
        float p = __expf(sf[nt][r] - mnew);
        sf[nt][r] = p;
        ps += p;
      }
#pragma unroll
      for (int d = 1; d < 16; d <<= 1) ps += __shfl_xor(ps, d, 32);
      rsum[r] = rsum[r] * al + ps;
#pragma unroll
      for (int dt = 0; dt < 4; ++dt) o[dt][r] *= al;
    }

    // stage P as bf16 in A-layout source tile
#pragma unroll
    for (int nt = 0; nt < 4; ++nt)
#pragma unroll
      for (int r = 0; r < 8; ++r)
        Ps[wave][hh * 8 + r][nt * 16 + nl] = f2bf(sf[nt][r]);
    __syncthreads();

    // O += P V : 4 dk-tiles x 2 key-chunks
#pragma unroll
    for (int dt = 0; dt < 4; ++dt)
#pragma unroll
      for (int kc = 0; kc < 2; ++kc) {
        v16us a  = frag_apairs(&Ps[wave][nl][kc * 32], hh);
        v16us bb = frag_bpairs(&Vst[dt * 16 + nl][kc * 32], hh);
        o[dt] = wmma_bf16(a, bb, o[dt]);
      }
  }

  // normalize, zero query row 0, store bf16
#pragma unroll
  for (int dt = 0; dt < 4; ++dt)
#pragma unroll
    for (int r = 0; r < 8; ++r) {
      float v = o[dt][r] / rsum[r];
      if (qrow[r] == 0) v = 0.0f;
      O[baseQ + (size_t)qrow[r] * Dc + dt * 16 + nl] = f2bf(v);
    }
}

// ---------------- fused residual + LayerNorm (one wave per 512-wide row) ----------------
__global__ __launch_bounds__(256) void ln_kernel(
    const float* __restrict__ xin, const float* __restrict__ res,
    const float* __restrict__ g, const float* __restrict__ be,
    float* __restrict__ xoutF, unsigned short* __restrict__ xoutB) {
  int row = blockIdx.x * 8 + (threadIdx.x >> 5);
  int lane = threadIdx.x & 31;
  size_t base = (size_t)row * Dc + lane * 16;
  float v[16];
  float s = 0.0f, s2 = 0.0f;
#pragma unroll
  for (int j = 0; j < 4; ++j) {
    float4 a = *(const float4*)(xin + base + j * 4);
    float4 b = *(const float4*)(res + base + j * 4);
    float t0 = a.x + b.x, t1 = a.y + b.y, t2 = a.z + b.z, t3 = a.w + b.w;
    v[j * 4 + 0] = t0; v[j * 4 + 1] = t1; v[j * 4 + 2] = t2; v[j * 4 + 3] = t3;
    s += t0 + t1 + t2 + t3;
    s2 += t0 * t0 + t1 * t1 + t2 * t2 + t3 * t3;
  }
#pragma unroll
  for (int d = 1; d < 32; d <<= 1) {
    s += __shfl_xor(s, d, 32);
    s2 += __shfl_xor(s2, d, 32);
  }
  float mean = s * (1.0f / 512.0f);
  float var = s2 * (1.0f / 512.0f) - mean * mean;
  float rstd = rsqrtf(var + 1e-5f);
  int c0 = lane * 16;
#pragma unroll
  for (int j = 0; j < 4; ++j) {
    float4 gg = *(const float4*)(g + c0 + j * 4);
    float4 bb = *(const float4*)(be + c0 + j * 4);
    float o0 = (v[j * 4 + 0] - mean) * rstd * gg.x + bb.x;
    float o1 = (v[j * 4 + 1] - mean) * rstd * gg.y + bb.y;
    float o2 = (v[j * 4 + 2] - mean) * rstd * gg.z + bb.z;
    float o3 = (v[j * 4 + 3] - mean) * rstd * gg.w + bb.w;
    *(float4*)(xoutF + base + j * 4) = make_float4(o0, o1, o2, o3);
    *(uint2*)(xoutB + base + j * 4) = make_uint2(pack2bf(o0, o1), pack2bf(o2, o3));
  }
}

__global__ __launch_bounds__(256) void copy_kernel(
    const float* __restrict__ in, float* __restrict__ out) {
  int i = blockIdx.x * 256 + threadIdx.x;
  ((float4*)out)[i] = ((const float4*)in)[i];
}

// ---------------- host launch ----------------
extern "C" void kernel_launch(void* const* d_in, const int* in_sizes, int n_in,
                              void* d_out, int out_size, void* d_ws, size_t ws_size,
                              hipStream_t stream) {
  (void)in_sizes; (void)n_in; (void)out_size; (void)ws_size;
  const float* qe  = (const float*)d_in[0];
  const float* qa  = (const float*)d_in[1];
  const float* fo  = (const float*)d_in[2];
  const float* pe  = (const float*)d_in[3];
  const float* Wk  = (const float*)d_in[4];
  const float* bk  = (const float*)d_in[5];
  const float* Wv  = (const float*)d_in[6];
  const float* bv  = (const float*)d_in[7];
  const float* Wo  = (const float*)d_in[8];
  const float* bo  = (const float*)d_in[9];
  const float* g1  = (const float*)d_in[10];
  const float* be1 = (const float*)d_in[11];
  const float* W1  = (const float*)d_in[12];
  const float* bf1 = (const float*)d_in[13];
  const float* W2  = (const float*)d_in[14];
  const float* bf2 = (const float*)d_in[15];
  const float* g2  = (const float*)d_in[16];
  const float* be2 = (const float*)d_in[17];

  char* wp = (char*)d_ws;
  auto alloc = [&](size_t bytes) -> char* {
    char* p = wp;
    wp += (bytes + 255) & ~(size_t)255;
    return p;
  };
  float* xf = (float*)alloc(sizeof(float) * Mrows * Dc);
  float* t0 = (float*)alloc(sizeof(float) * Mrows * Dc);
  unsigned short* xb  = (unsigned short*)alloc((size_t)2 * Mrows * Dc);
  unsigned short* yb  = (unsigned short*)alloc((size_t)2 * Mrows * Dc);
  unsigned short* qb  = (unsigned short*)alloc((size_t)2 * Mrows * Dc);
  unsigned short* vb  = (unsigned short*)alloc((size_t)2 * Mrows * Dc);
  unsigned short* ab  = (unsigned short*)alloc((size_t)2 * Mrows * Dc);
  unsigned short* hb  = (unsigned short*)alloc((size_t)2 * Mrows * FFc);
  unsigned short* wkb = (unsigned short*)alloc((size_t)2 * NBc * Dc * Dc);
  unsigned short* wvb = (unsigned short*)alloc((size_t)2 * NBc * Dc * Dc);
  unsigned short* wob = (unsigned short*)alloc((size_t)2 * NBc * Dc * Dc);
  unsigned short* w1b = (unsigned short*)alloc((size_t)2 * NBc * Dc * FFc);
  unsigned short* w2b = (unsigned short*)alloc((size_t)2 * NBc * FFc * Dc);

  prep_kernel<<<Mrows * Dc / 4 / 256, 256, 0, stream>>>(qe, qa, pe, xf, xb, yb);
  cvt_bf16_kernel<<<NBc * Dc * Dc / 4 / 256, 256, 0, stream>>>(Wk, wkb);
  cvt_bf16_kernel<<<NBc * Dc * Dc / 4 / 256, 256, 0, stream>>>(Wv, wvb);
  cvt_bf16_kernel<<<NBc * Dc * Dc / 4 / 256, 256, 0, stream>>>(Wo, wob);
  cvt_bf16_kernel<<<NBc * Dc * FFc / 4 / 256, 256, 0, stream>>>(W1, w1b);
  cvt_bf16_kernel<<<NBc * FFc * Dc / 4 / 256, 256, 0, stream>>>(W2, w2b);

  dim3 gD(Dc / 64, Mrows / 128);
  dim3 gF(FFc / 64, Mrows / 128);
  for (int i = 0; i < NBc; ++i) {
    gemm_kernel<<<gD, 256, 0, stream>>>(xb, wkb + (size_t)i * Dc * Dc, bk + i * Dc,
                                        nullptr, qb, 0, Mrows, Dc, Dc);
    gemm_kernel<<<gD, 256, 0, stream>>>(yb, wvb + (size_t)i * Dc * Dc, bv + i * Dc,
                                        nullptr, vb, 0, Mrows, Dc, Dc);
    attn_kernel<<<dim3(Bc * Hc, Sc / 64), 128, 0, stream>>>(qb, vb, fo, ab);
    gemm_kernel<<<gD, 256, 0, stream>>>(ab, wob + (size_t)i * Dc * Dc, bo + i * Dc,
                                        t0, nullptr, 0, Mrows, Dc, Dc);
    ln_kernel<<<Mrows / 8, 256, 0, stream>>>(xf, t0, g1 + i * Dc, be1 + i * Dc, xf, xb);
    gemm_kernel<<<gF, 256, 0, stream>>>(xb, w1b + (size_t)i * Dc * FFc, bf1 + i * FFc,
                                        nullptr, hb, 1, Mrows, FFc, Dc);
    gemm_kernel<<<gD, 256, 0, stream>>>(hb, w2b + (size_t)i * FFc * Dc, bf2 + i * Dc,
                                        t0, nullptr, 0, Mrows, Dc, FFc);
    ln_kernel<<<Mrows / 8, 256, 0, stream>>>(xf, t0, g2 + i * Dc, be2 + i * Dc, xf, xb);
  }
  copy_kernel<<<Mrows * Dc / 4 / 256, 256, 0, stream>>>(xf, (float*)d_out);
}
